// BiAM_88888643158088
// MI455X (gfx1250) — matrix operand order repaired and checked by reading, not compiled, not run on hardware
//
#include <hip/hip_runtime.h>
#include <math.h>

typedef __attribute__((ext_vector_type(16))) _Float16 v16h;
typedef __attribute__((ext_vector_type(8)))  _Float16 v8h;
typedef __attribute__((ext_vector_type(8)))  float    v8f;

#define A_ROW    0
#define A_NCHW   1
#define A_CONV3  2
#define A_CONVBP 3
#define A_DECONV 4
#define A_HEADQ  5
#define B_W16    0
#define B_HEADK  1
#define B_HEADV  2
#define O_ROW    0
#define O_NCHW   1
#define O_HEAD   2
#define ACT_NONE  0
#define ACT_RELU  1
#define ACT_LRELU 2

struct GP {
  const float*    A;
  const _Float16* Bw;     // f16 weights, [N][K] layout
  const float*    Bf;     // fp32 gather source for attention B operands
  float*          C;
  const float*    bias;   // always valid (dummy zeros when unused)
  const float*    bnS;    // always valid (dummy ones when unused)
  const float*    bnB;    // always valid (dummy zeros when unused)
  const float*    res;
  float alpha;
  int M, N, K;
  int lda;
  int bA, bC;             // per-z element offsets (batched row-major modes)
  int S, Cin, Hh, Ww;     // A-side NCHW / conv geometry
  int Sout, Ctot, chOff;  // output NCHW geometry
};

// ---------------- A gather: per-thread prep (loop-invariant) + per-k element ----------------
struct AP { int a, b, c, d; bool ok; };

template<int AM>
__device__ __forceinline__ AP prepA(const GP& p, int m, int z) {
  AP r; r.ok = m < p.M; int mm = r.ok ? m : 0;
  r.a = 0; r.b = 0; r.c = 0; r.d = 0;
  if constexpr (AM == A_ROW) {
    r.a = mm * p.lda;
  } else if constexpr (AM == A_NCHW) {
    int bb = mm / p.S; int s = mm - bb * p.S;
    r.a = bb * p.Cin * p.S + s;
  } else if constexpr (AM == A_CONV3) {
    int bb = mm / p.S; int s = mm - bb * p.S;
    r.a = bb * p.Cin * p.S;
    r.b = s / p.Ww; r.c = s - r.b * p.Ww;
  } else if constexpr (AM == A_CONVBP) {
    // feature_map (128,512,14,14) viewed as 2x2 blocks of 7x7; img=(pB*2+qB)*128+b
    int img = mm / 49; int s = mm - img * 49;
    int pB = img >> 8, qB = (img >> 7) & 1, bb = img & 127;
    r.a = bb * 512 * 196;
    r.b = s / 7; r.c = s - r.b * 7;
    r.d = pB * 98 + qB * 7;               // block-corner spatial offset
  } else if constexpr (AM == A_DECONV) {
    int img = mm / 196; int s = mm - img * 196;
    r.a = img * 256 * 49;
    r.b = s / 14; r.c = s - r.b * 14;     // oy, ox
  } else { // A_HEADQ: A[m,k] = q[b][h*32+k][m]
    int bb = z >> 3, h = z & 7;
    r.a = (bb * 256 + h * 32) * 196 + mm;
  }
  return r;
}

template<int AM>
__device__ __forceinline__ int elemA(const GP& p, const AP& r, int k, bool& ok) {
  if constexpr (AM == A_ROW) {
    ok = r.ok && (k < p.K);
    return ok ? (r.a + k) : 0;
  } else if constexpr (AM == A_NCHW) {
    ok = r.ok;
    return ok ? (r.a + k * p.S) : 0;
  } else if constexpr (AM == A_CONV3) {
    int c = k / 9, t = k - 9 * c;
    int yy = r.b + t / 3 - 1, xx = r.c + t % 3 - 1;
    ok = r.ok && (unsigned)yy < (unsigned)p.Hh && (unsigned)xx < (unsigned)p.Ww;
    return ok ? (r.a + c * p.S + yy * p.Ww + xx) : 0;
  } else if constexpr (AM == A_CONVBP) {
    int c = k / 9, t = k - 9 * c;
    int yy = r.b + t / 3 - 1, xx = r.c + t % 3 - 1;
    ok = r.ok && (unsigned)yy < 7u && (unsigned)xx < 7u;
    return ok ? (r.a + c * 196 + r.d + yy * 14 + xx) : 0;
  } else if constexpr (AM == A_DECONV) {
    // ConvTranspose2d k3 s2 p1 op1: oy = 2*iy - 1 + ky
    int i = k / 9, t = k - 9 * i;
    int ky = t / 3, kx = t - 3 * ky;
    int ny = r.b + 1 - ky, nx = r.c + 1 - kx;
    ok = r.ok && ny >= 0 && nx >= 0 && !(ny & 1) && !(nx & 1) && (ny >> 1) < 7 && (nx >> 1) < 7;
    return ok ? (r.a + i * 49 + (ny >> 1) * 7 + (nx >> 1)) : 0;
  } else { // A_HEADQ
    ok = r.ok;
    return ok ? (r.a + k * 196) : 0;
  }
}

// ---------------- B gather prep/elem ----------------
struct BPr { int a; bool ok; };

template<int BM>
__device__ __forceinline__ BPr prepB(const GP& p, int n, int z) {
  BPr r; r.ok = n < p.N; int nn = r.ok ? n : 0;
  if constexpr (BM == B_HEADK) {        // B[k,n] = K[b][h*32+k][n]
    int bb = z >> 3, h = z & 7;
    r.a = (bb * 256 + h * 32) * 196 + nn;
  } else if constexpr (BM == B_HEADV) { // B[k,n] = V[b][h*32+n][k]
    int bb = z >> 3, h = z & 7;
    r.a = (bb * 256 + h * 32 + nn) * 196;
  } else {                              // B_W16: row base in [N][K] layout
    r.a = nn * p.K;
  }
  return r;
}

template<int BM>
__device__ __forceinline__ int elemB(const GP& p, const BPr& r, int k, bool& ok) {
  if constexpr (BM == B_HEADK) { ok = r.ok; return ok ? (r.a + k * 196) : 0; }
  else                         { ok = r.ok && (k < p.K); return ok ? (r.a + k) : 0; }
}

// ---------------- gather one K-step worth of A/B fragments into registers ----------------
template<int AM, int BM>
__device__ __forceinline__ void gatherAB(const GP& p, const float* Az, const AP& ra, const BPr& rb,
                                         int abk, int bkq, int k0,
                                         v8h& pa0, v8h& pa1, v8h& pb) {
  int ix0[8], ix1[8]; bool ok0[8], ok1[8];
  #pragma unroll
  for (int j = 0; j < 8; j++) ix0[j] = elemA<AM>(p, ra, k0 + abk + j, ok0[j]);
  #pragma unroll
  for (int j = 0; j < 8; j++) ix1[j] = elemA<AM>(p, ra, k0 + abk + 8 + j, ok1[j]);
  float f0[8], f1[8];
  #pragma unroll
  for (int j = 0; j < 8; j++) f0[j] = Az[ix0[j]];
  #pragma unroll
  for (int j = 0; j < 8; j++) f1[j] = Az[ix1[j]];
  #pragma unroll
  for (int j = 0; j < 8; j++) pa0[j] = ok0[j] ? (_Float16)f0[j] : (_Float16)0.f;
  #pragma unroll
  for (int j = 0; j < 8; j++) pa1[j] = ok1[j] ? (_Float16)f1[j] : (_Float16)0.f;
  if constexpr (BM == B_W16) {
    pb = *(const v8h*)(p.Bw + rb.a + k0 + bkq);   // row 0 is always a safe address
    if (!rb.ok) { v8h zv = {}; pb = zv; }
  } else {
    int ixb[8]; bool okb[8]; float fb[8];
    #pragma unroll
    for (int j = 0; j < 8; j++) ixb[j] = elemB<BM>(p, rb, k0 + bkq + j, okb[j]);
    #pragma unroll
    for (int j = 0; j < 8; j++) fb[j] = p.Bf[ixb[j]];
    #pragma unroll
    for (int j = 0; j < 8; j++) pb[j] = okb[j] ? (_Float16)fb[j] : (_Float16)0.f;
  }
}

// ---------------- tiled WMMA GEMM: 128x64 macro tile, 8 waves, K-step 32 ----------------
// Double-buffered LDS, one barrier per K-step, next-step gathers overlap WMMA compute.
template<int AM, int BM, int OM, int ACT, bool RES>
__global__ __launch_bounds__(256) void gemm_wmma(GP p) {
  __shared__ _Float16 As[2 * 128 * 40];
  __shared__ _Float16 Bs[2 * 64 * 40];
  const int tid = threadIdx.x;
  const int z   = blockIdx.z;
  const int tM  = blockIdx.y * 128;
  const int tN  = blockIdx.x * 64;
  const float* Az = p.A + (long long)z * p.bA;
  const int lane = tid & 31, wv = tid >> 5;
  const int wm = wv & 3, wn = wv >> 2;          // wave grid 4(M) x 2(N)
  const int hi = lane >> 4, lm = lane & 15;
  const int arow = tid >> 1, abk = (tid & 1) * 16;
  const int brow = tid >> 2, bkq = (tid & 3) * 8;

  const AP  ra = prepA<AM>(p, tM + arow, z);
  const BPr rb = prepB<BM>(p, tN + brow, z);

  v8f acc[2][2];
  #pragma unroll
  for (int i = 0; i < 2; i++)
    #pragma unroll
    for (int j = 0; j < 2; j++) { v8f zz = {}; acc[i][j] = zz; }

  v8h pa0 = {}, pa1 = {}, pb = {};
  gatherAB<AM, BM>(p, Az, ra, rb, abk, bkq, 0, pa0, pa1, pb);

  for (int k0 = 0; k0 < p.K; k0 += 32) {
    _Float16* Asb = As + ((k0 >> 5) & 1) * (128 * 40);
    _Float16* Bsb = Bs + ((k0 >> 5) & 1) * (64 * 40);
    *(v8h*)&Asb[arow * 40 + abk]     = pa0;
    *(v8h*)&Asb[arow * 40 + abk + 8] = pa1;
    *(v8h*)&Bsb[brow * 40 + bkq]     = pb;
    if (k0 + 32 < p.K)   // prefetch next K-step; loads fly during the WMMAs below
      gatherAB<AM, BM>(p, Az, ra, rb, abk, bkq, k0 + 32, pa0, pa1, pb);
    __syncthreads();

    union HV { v16h v; v8h h[2]; };
    HV af[2], bf[2];
    #pragma unroll
    for (int i = 0; i < 2; i++) {
      int row = wm * 32 + i * 16 + lm;
      // A layout: halves 0..15 -> K=(hi?16:0)+0..15 for M=lm (matches B)
      af[i].h[0] = *(const v8h*)&Asb[row * 40 + hi * 16];
      af[i].h[1] = *(const v8h*)&Asb[row * 40 + hi * 16 + 8];
    }
    #pragma unroll
    for (int j = 0; j < 2; j++) {
      int col = wn * 32 + j * 16 + lm;
      // B layout, same k order as A: halves 0..15 -> K=(hi?16:0)+0..15 for N=lm
      bf[j].h[0] = *(const v8h*)&Bsb[col * 40 + hi * 16];
      bf[j].h[1] = *(const v8h*)&Bsb[col * 40 + hi * 16 + 8];
    }
    #pragma unroll
    for (int i = 0; i < 2; i++)
      #pragma unroll
      for (int j = 0; j < 2; j++)
        acc[i][j] = __builtin_amdgcn_wmma_f32_16x16x32_f16(
            false, af[i].v, false, bf[j].v, (short)0, acc[i][j], false, false);
  }

  // ---------------- fused epilogue (fully branchless scalar path) ----------------
  int zo = z * p.bC;
  #pragma unroll
  for (int i = 0; i < 2; i++) {
    #pragma unroll
    for (int j = 0; j < 2; j++) {
      int ng = tN + wn * 32 + j * 16 + lm;
      if (ng >= p.N) continue;
      float bia = p.bias[ng];
      float s1  = p.bnS[ng];
      float s0  = p.bnB[ng];
      #pragma unroll
      for (int r = 0; r < 8; r++) {
        int mg = tM + wm * 32 + i * 16 + hi * 8 + r;  // ISA C layout: M = r + 8*hi
        if (mg >= p.M) continue;
        float v = acc[i][j][r] * p.alpha + bia;
        if constexpr (ACT == ACT_RELU)  v = fmaxf(v, 0.f);
        if constexpr (ACT == ACT_LRELU) v = (v > 0.f) ? v : 0.2f * v;
        v = v * s1 + s0;
        int idx;
        if constexpr (OM == O_ROW) {
          idx = zo + mg * p.N + ng;
        } else if constexpr (OM == O_NCHW) {
          int b = mg / p.Sout; int s = mg - b * p.Sout;
          idx = (b * p.Ctot + p.chOff + ng) * p.Sout + s;
        } else { // O_HEAD: out[b][h*32+n][m]
          int b = z >> 3, h = z & 7;
          idx = (b * 256 + h * 32 + ng) * 196 + mg;
        }
        if constexpr (RES) v += p.res[idx];
        p.C[idx] = v;
      }
    }
  }
}

template<int AM, int BM, int OM, int ACT, bool RES>
static void gemm_go(const GP& p, int Z, hipStream_t s) {
  dim3 grid((p.N + 63) / 64, (p.M + 127) / 128, (unsigned)Z);
  gemm_wmma<AM, BM, OM, ACT, RES><<<grid, dim3(256), 0, s>>>(p);
}

// 1x1 conv helper (compile-time ACT/RES)
template<int ACT, bool RES>
static void c1go(const float* in, int Cin, const _Float16* W, const float* bias, int Cout,
                 float* out, int Ctot, int chOff, const float* res,
                 const float* don, const float* dz, hipStream_t stream) {
  GP g{}; g.A = in; g.Bw = W; g.C = out; g.bias = bias; g.bnS = don; g.bnB = dz; g.res = res;
  g.alpha = 1.f; g.M = 128 * 196; g.N = Cout; g.K = Cin; g.S = 196; g.Cin = Cin;
  g.Sout = 196; g.Ctot = Ctot; g.chOff = chOff;
  gemm_go<A_NCHW, B_W16, O_NCHW, ACT, RES>(g, 1, stream);
}

// ---------------- small kernels ----------------
__global__ void k_cvt(const float* w, _Float16* o, long long total) {
  long long i = (long long)blockIdx.x * 256 + threadIdx.x;
  if (i >= total) return;
  o[i] = (_Float16)w[i];
}
// deconv IOHW (256,256,3,3) -> [N=o][K=i*9+t] f16
__global__ void k_dcw(const float* w, _Float16* o) {
  int j = blockIdx.x * 256 + threadIdx.x;
  if (j >= 256 * 2304) return;
  int oc = j / 2304, rem = j - oc * 2304;
  int i = rem / 9, t = rem - i * 9;
  o[j] = (_Float16)w[(i * 256 + oc) * 9 + t];
}
__global__ void k_fill(float* p, float v, int n) {
  int i = blockIdx.x * 256 + threadIdx.x;
  if (i < n) p[i] = v;
}
__global__ void k_bnprep(const float* g, const float* bt, const float* m, const float* v,
                         float* sc, float* sh, int C) {
  int i = threadIdx.x;
  if (i < C) { float s = g[i] * rsqrtf(v[i] + 1e-3f); sc[i] = s; sh[i] = bt[i] - m[i] * s; }
}
__global__ void k_max4(const float* d, float* o, int n, int off) {
  int i = blockIdx.x * 256 + threadIdx.x;
  if (i >= n) return;
  float a = fmaxf(d[i], d[i + off]);
  float b = fmaxf(d[i + 2 * off], d[i + 3 * off]);
  o[i] = fmaxf(a, b);
}
// masked softmax: scores==0 -> -1e9 then softmax over last axis (196)
__global__ void k_softmax(float* sc) {
  int wid = threadIdx.x >> 5, lane = threadIdx.x & 31;
  int row = blockIdx.y * 196 + blockIdx.x * 4 + wid;
  float* r = sc + row * 196;
  float v[7]; float mx = -1e30f;
  #pragma unroll
  for (int i2 = 0; i2 < 7; i2++) {
    int c = lane + i2 * 32; float x = -1e30f;
    if (c < 196) { x = r[c]; if (x == 0.f) x = -1e9f; }
    v[i2] = x; mx = fmaxf(mx, x);
  }
  for (int o = 16; o; o >>= 1) mx = fmaxf(mx, __shfl_xor(mx, o, 32));
  float sum = 0.f;
  #pragma unroll
  for (int i2 = 0; i2 < 7; i2++) {
    int c = lane + i2 * 32;
    if (c < 196) { float e = __expf(v[i2] - mx); v[i2] = e; sum += e; }
  }
  for (int o = 16; o; o >>= 1) sum += __shfl_xor(sum, o, 32);
  float inv = 1.f / sum;
  #pragma unroll
  for (int i2 = 0; i2 < 7; i2++) { int c = lane + i2 * 32; if (c < 196) r[c] = v[i2] * inv; }
}
// r_g = sigmoid(mean_s(q_g) * k_g); c_g = r_g * q_g   (one wave per (b,c))
__global__ void k_rgcg(const float* qg, const float* kg, float* cg) {
  int wid = threadIdx.x >> 5, lane = threadIdx.x & 31;
  int bc = blockIdx.x * 4 + wid;
  const float* src = qg + bc * 196; float* dst = cg + bc * 196;
  float s = 0.f;
  for (int i2 = 0; i2 < 7; i2++) { int c = lane + i2 * 32; if (c < 196) s += src[c]; }
  for (int o = 16; o; o >>= 1) s += __shfl_xor(s, o, 32);
  float mean = s * (1.f / 196.f);
  float rg = 1.f / (1.f + __expf(-mean * kg[bc]));
  for (int i2 = 0; i2 < 7; i2++) { int c = lane + i2 * 32; if (c < 196) dst[c] = rg * src[c]; }
}
__global__ void k_pool(const float* e, float* o) {
  int wid = threadIdx.x >> 5, lane = threadIdx.x & 31;
  int bc = blockIdx.x * 4 + wid;
  const float* src = e + bc * 196;
  float s = 0.f;
  for (int i2 = 0; i2 < 7; i2++) { int c = lane + i2 * 32; if (c < 196) s += src[c]; }
  for (int o2 = 16; o2; o2 >>= 1) s += __shfl_xor(s, o2, 32);
  if (lane == 0) o[bc] = s * (1.f / 196.f);
}

// ---------------- driver ----------------
extern "C" void kernel_launch(void* const* d_in, const int* in_sizes, int n_in,
                              void* d_out, int out_size, void* d_ws, size_t ws_size,
                              hipStream_t stream) {
  (void)in_sizes; (void)n_in; (void)out_size; (void)ws_size;
  const float* fm    = (const float*)d_in[0];
  const float* x_g   = (const float*)d_in[1];
  const float* bpc_w = (const float*)d_in[2];  const float* bpc_b = (const float*)d_in[3];
  const float* dc_w  = (const float*)d_in[4];  const float* dc_b  = (const float*)d_in[5];
  const float* c33_w = (const float*)d_in[6];  const float* c33_b = (const float*)d_in[7];
  const float* c33_g = (const float*)d_in[8];  const float* c33_bt= (const float*)d_in[9];
  const float* c33_m = (const float*)d_in[10]; const float* c33_v = (const float*)d_in[11];
  const float* wq_w  = (const float*)d_in[12]; const float* wq_b  = (const float*)d_in[13];
  const float* wk_w  = (const float*)d_in[14]; const float* wk_b  = (const float*)d_in[15];
  const float* wv_w  = (const float*)d_in[16]; const float* wv_b  = (const float*)d_in[17];
  const float* wo_w  = (const float*)d_in[18]; const float* wo_b  = (const float*)d_in[19];
  const float* ff1_w = (const float*)d_in[20]; const float* ff1_b = (const float*)d_in[21];
  const float* ff2_w = (const float*)d_in[22]; const float* ff2_b = (const float*)d_in[23];
  const float* cc_w  = (const float*)d_in[24]; const float* cc_b  = (const float*)d_in[25];
  const float* wg_w  = (const float*)d_in[26]; const float* wg_b  = (const float*)d_in[27];
  const float* gc_w  = (const float*)d_in[28]; const float* gc_b  = (const float*)d_in[29];
  const float* gc_g  = (const float*)d_in[30]; const float* gc_bt = (const float*)d_in[31];
  const float* gc_m  = (const float*)d_in[32]; const float* gc_v  = (const float*)d_in[33];
  const float* saq_w = (const float*)d_in[34]; const float* saq_b = (const float*)d_in[35];
  const float* sak_w = (const float*)d_in[36]; const float* sak_b = (const float*)d_in[37];
  const float* sav_w = (const float*)d_in[38]; const float* sav_b = (const float*)d_in[39];
  const float* sao_w = (const float*)d_in[40]; const float* sao_b = (const float*)d_in[41];
  const float* f11w  = (const float*)d_in[42]; const float* f11b  = (const float*)d_in[43];
  const float* fcw   = (const float*)d_in[44]; const float* fcb   = (const float*)d_in[45];

  // ---- workspace carve (deterministic every call) ----
  size_t off = 0;
  auto alloc = [&](size_t bytes) -> void* {
    void* r = (char*)d_ws + off;
    off += (bytes + 255) & ~(size_t)255;
    return r;
  };
  _Float16* bpW  = (_Float16*)alloc(256LL * 4608 * 2);
  _Float16* dcW  = (_Float16*)alloc(256LL * 2304 * 2);
  _Float16* c33W = (_Float16*)alloc(256LL * 2304 * 2);
  _Float16* gcW  = (_Float16*)alloc(256LL * 2304 * 2);
  _Float16* wqW  = (_Float16*)alloc(256 * 256 * 2);
  _Float16* wkW  = (_Float16*)alloc(256 * 256 * 2);
  _Float16* wvW  = (_Float16*)alloc(256 * 256 * 2);
  _Float16* woW  = (_Float16*)alloc(256 * 256 * 2);
  _Float16* ccW  = (_Float16*)alloc(256 * 256 * 2);
  _Float16* saqW = (_Float16*)alloc(256 * 256 * 2);
  _Float16* sakW = (_Float16*)alloc(256 * 256 * 2);
  _Float16* savW = (_Float16*)alloc(256 * 256 * 2);
  _Float16* saoW = (_Float16*)alloc(256 * 256 * 2);
  _Float16* ff1W = (_Float16*)alloc(512 * 256 * 2);   // [N=512][K=256]
  _Float16* ff2W = (_Float16*)alloc(256 * 512 * 2);   // [N=256][K=512]
  _Float16* wgW  = (_Float16*)alloc(256 * 1024 * 2);  // [N=256][K=1024]
  _Float16* f11W = (_Float16*)alloc(512 * 512 * 2);
  _Float16* fcW  = (_Float16*)alloc(636 * 512 * 2);   // [N=636][K=512]
  float* c33s  = (float*)alloc(256 * 4); float* c33sh = (float*)alloc(256 * 4);
  float* gcs   = (float*)alloc(256 * 4); float* gcsh  = (float*)alloc(256 * 4);
  float* dz    = (float*)alloc(640 * 4);   // dummy zeros (bias / bn shift)
  float* don   = (float*)alloc(640 * 4);   // dummy ones  (bn scale)
  float* kgB   = (float*)alloc(128 * 256 * 4);
  float* pooled= (float*)alloc(128 * 512 * 4);
  // big region: y1 (25.7MB) + dec (102.8MB) early, aliased later by scores (157.4MB)
  char*  big   = (char*)alloc(157351936);
  float* y1    = (float*)big;
  float* dec   = (float*)(big + 25690112);
  float* scores= (float*)big;
  const size_t CSb = 25690112;            // 128*256*196 fp32
  float* x_r  = (float*)alloc(CSb);
  float* h_r  = (float*)alloc(CSb);
  float* qb   = (float*)alloc(CSb);
  float* kb   = (float*)alloc(CSb);
  float* vb   = (float*)alloc(CSb);
  float* mhab = (float*)alloc(CSb);
  float* o_r  = (float*)alloc(CSb);
  float* e_r  = (float*)alloc(CSb);
  float* q_g  = (float*)alloc(CSb);
  float* c_g  = (float*)alloc(CSb);
  float* e_g  = x_r;                       // x_r dead after CONV3_3
  float* ffh  = (float*)alloc(51380224);   // 128*512*196 fp32
  float* ef_in= ffh;                       // ffh dead before concat buffer is written
  float* e_f  = (float*)alloc(51380224);

  // ---- weight prep (fp32 -> f16, [N][K] layouts; OIHW / [O,I] are already [N][K]) ----
  k_cvt<<<(4608 * 256 + 255) / 256, 256, 0, stream>>>(bpc_w, bpW, 4608LL * 256);
  k_dcw<<<(2304 * 256 + 255) / 256, 256, 0, stream>>>(dc_w, dcW);
  k_cvt<<<(2304 * 256 + 255) / 256, 256, 0, stream>>>(c33_w, c33W, 2304LL * 256);
  k_cvt<<<(2304 * 256 + 255) / 256, 256, 0, stream>>>(gc_w, gcW, 2304LL * 256);
  k_cvt<<<(256 * 256 + 255) / 256, 256, 0, stream>>>(wq_w, wqW, 256LL * 256);
  k_cvt<<<(256 * 256 + 255) / 256, 256, 0, stream>>>(wk_w, wkW, 256LL * 256);
  k_cvt<<<(256 * 256 + 255) / 256, 256, 0, stream>>>(wv_w, wvW, 256LL * 256);
  k_cvt<<<(256 * 256 + 255) / 256, 256, 0, stream>>>(wo_w, woW, 256LL * 256);
  k_cvt<<<(256 * 256 + 255) / 256, 256, 0, stream>>>(cc_w, ccW, 256LL * 256);
  k_cvt<<<(256 * 256 + 255) / 256, 256, 0, stream>>>(saq_w, saqW, 256LL * 256);
  k_cvt<<<(256 * 256 + 255) / 256, 256, 0, stream>>>(sak_w, sakW, 256LL * 256);
  k_cvt<<<(256 * 256 + 255) / 256, 256, 0, stream>>>(sav_w, savW, 256LL * 256);
  k_cvt<<<(256 * 256 + 255) / 256, 256, 0, stream>>>(sao_w, saoW, 256LL * 256);
  k_cvt<<<(512 * 256 + 255) / 256, 256, 0, stream>>>(ff1_w, ff1W, 512LL * 256);
  k_cvt<<<(512 * 256 + 255) / 256, 256, 0, stream>>>(ff2_w, ff2W, 512LL * 256);
  k_cvt<<<(1024 * 256 + 255) / 256, 256, 0, stream>>>(wg_w, wgW, 1024LL * 256);
  k_cvt<<<(512 * 512 + 255) / 256, 256, 0, stream>>>(f11w, f11W, 512LL * 512);
  k_cvt<<<(636 * 512 + 255) / 256, 256, 0, stream>>>(fcw, fcW, 636LL * 512);
  k_fill<<<3, 256, 0, stream>>>(dz, 0.f, 640);
  k_fill<<<3, 256, 0, stream>>>(don, 1.f, 640);
  k_bnprep<<<1, 256, 0, stream>>>(c33_g, c33_bt, c33_m, c33_v, c33s, c33sh, 256);
  k_bnprep<<<1, 256, 0, stream>>>(gc_g, gc_bt, gc_m, gc_v, gcs, gcsh, 256);

  // ---- attention helper ----
  auto attn = [&](const float* q, const float* k, const float* v, float* outb) {
    GP g{}; g.A = q; g.Bf = k; g.C = scores; g.bias = dz; g.bnS = don; g.bnB = dz;
    g.alpha = 0.17677669529663687f; // 1/sqrt(32)
    g.M = 196; g.N = 196; g.K = 32; g.bA = 0; g.bC = 196 * 196;
    gemm_go<A_HEADQ, B_HEADK, O_ROW, ACT_NONE, false>(g, 1024, stream);
    k_softmax<<<dim3(49, 1024), 128, 0, stream>>>(scores);
    GP g2{}; g2.A = scores; g2.lda = 196; g2.bA = 196 * 196; g2.Bf = v; g2.C = outb;
    g2.bias = dz; g2.bnS = don; g2.bnB = dz; g2.alpha = 1.f;
    g2.M = 196; g2.N = 32; g2.K = 196;
    gemm_go<A_ROW, B_HEADV, O_HEAD, ACT_NONE, false>(g2, 1024, stream);
  };

  // 1) BlockProcessor conv3x3 512->256 on split blocks (+relu)
  { GP g{}; g.A = fm; g.Bw = bpW; g.C = y1; g.bias = bpc_b; g.bnS = don; g.bnB = dz;
    g.alpha = 1.f; g.M = 512 * 49; g.N = 256; g.K = 4608; g.Sout = 49; g.Ctot = 256; g.chOff = 0;
    gemm_go<A_CONVBP, B_W16, O_NCHW, ACT_RELU, false>(g, 1, stream); }
  // 2) deconv 7x7 -> 14x14 (+relu)
  { GP g{}; g.A = y1; g.Bw = dcW; g.C = dec; g.bias = dc_b; g.bnS = don; g.bnB = dz;
    g.alpha = 1.f; g.M = 512 * 196; g.N = 256; g.K = 2304; g.Sout = 196; g.Ctot = 256; g.chOff = 0;
    gemm_go<A_DECONV, B_W16, O_NCHW, ACT_RELU, false>(g, 1, stream); }
  // 3) max over 4 blocks -> x_r
  k_max4<<<(6422528 + 255) / 256, 256, 0, stream>>>(dec, x_r, 6422528, 6422528);
  // 4) CONV3_3: conv+relu+BN -> h_r
  { GP g{}; g.A = x_r; g.Bw = c33W; g.C = h_r; g.bias = c33_b; g.bnS = c33s; g.bnB = c33sh;
    g.alpha = 1.f; g.M = 25088; g.N = 256; g.K = 2304; g.S = 196; g.Cin = 256; g.Hh = 14; g.Ww = 14;
    g.Sout = 196; g.Ctot = 256; g.chOff = 0;
    gemm_go<A_CONV3, B_W16, O_NCHW, ACT_RELU, false>(g, 1, stream); }
  // 5) RCB self-attention
  c1go<ACT_NONE, false>(h_r, 256, wqW, wq_b, 256, qb, 256, 0, nullptr, don, dz, stream);
  c1go<ACT_NONE, false>(h_r, 256, wkW, wk_b, 256, kb, 256, 0, nullptr, don, dz, stream);
  c1go<ACT_NONE, false>(h_r, 256, wvW, wv_b, 256, vb, 256, 0, nullptr, don, dz, stream);
  attn(qb, kb, vb, mhab);
  c1go<ACT_NONE, true >(mhab, 256, woW, wo_b, 256, o_r, 256, 0, h_r, don, dz, stream);
  c1go<ACT_RELU, false>(o_r, 256, ff1W, ff1_b, 512, ffh, 512, 0, nullptr, don, dz, stream);
  c1go<ACT_NONE, true >(ffh, 512, ff2W, ff2_b, 256, e_r, 256, 0, o_r, don, dz, stream);
  // 6) SCB
  c1go<ACT_LRELU, false>(h_r, 256, ccW, cc_b, 256, q_g, 256, 0, nullptr, don, dz, stream);
  { GP g{}; g.A = x_g; g.lda = 1024; g.Bw = wgW; g.C = kgB; g.bias = wg_b; g.bnS = don; g.bnB = dz;
    g.alpha = 1.f; g.M = 128; g.N = 256; g.K = 1024;
    gemm_go<A_ROW, B_W16, O_ROW, ACT_NONE, false>(g, 1, stream); }
  k_rgcg<<<8192, 128, 0, stream>>>(q_g, kgB, c_g);
  { GP g{}; g.A = c_g; g.Bw = gcW; g.C = e_g; g.bias = gc_b; g.bnS = gcs; g.bnB = gcsh; g.res = c_g;
    g.alpha = 1.f; g.M = 25088; g.N = 256; g.K = 2304; g.S = 196; g.Cin = 256; g.Hh = 14; g.Ww = 14;
    g.Sout = 196; g.Ctot = 256; g.chOff = 0;
    gemm_go<A_CONV3, B_W16, O_NCHW, ACT_RELU, true>(g, 1, stream); }
  // 7) cross attention, branch r: q(e_r), k(e_g), v(e_g)
  c1go<ACT_NONE, false>(e_r, 256, saqW, saq_b, 256, qb, 256, 0, nullptr, don, dz, stream);
  c1go<ACT_NONE, false>(e_g, 256, sakW, sak_b, 256, kb, 256, 0, nullptr, don, dz, stream);
  c1go<ACT_NONE, false>(e_g, 256, savW, sav_b, 256, vb, 256, 0, nullptr, don, dz, stream);
  attn(qb, kb, vb, mhab);
  c1go<ACT_NONE, false>(mhab, 256, saoW, sao_b, 256, ef_in, 512, 0, nullptr, don, dz, stream);
  // 8) cross attention, branch g: q(e_g), k(e_r), v(e_r)
  c1go<ACT_NONE, false>(e_g, 256, saqW, saq_b, 256, qb, 256, 0, nullptr, don, dz, stream);
  c1go<ACT_NONE, false>(e_r, 256, sakW, sak_b, 256, kb, 256, 0, nullptr, don, dz, stream);
  c1go<ACT_NONE, false>(e_r, 256, savW, sav_b, 256, vb, 256, 0, nullptr, don, dz, stream);
  attn(qb, kb, vb, mhab);
  c1go<ACT_NONE, false>(mhab, 256, saoW, sao_b, 256, ef_in, 512, 256, nullptr, don, dz, stream);
  // 9) f11 1x1 512->512 + leaky relu
  c1go<ACT_LRELU, false>(ef_in, 512, f11W, f11b, 512, e_f, 512, 0, nullptr, don, dz, stream);
  // 10) global average pool
  k_pool<<<16384, 128, 0, stream>>>(e_f, pooled);
  // 11) fc 512->636 -> d_out
  { GP g{}; g.A = pooled; g.lda = 512; g.Bw = fcW; g.C = (float*)d_out; g.bias = fcb;
    g.bnS = don; g.bnB = dz; g.alpha = 1.f; g.M = 128; g.N = 636; g.K = 512;
    gemm_go<A_ROW, B_W16, O_ROW, ACT_NONE, false>(g, 1, stream); }
}
